// co_attentional_9715216024114
// MI455X (gfx1250) — matrix operand order repaired and checked by reading, not compiled, not run on hardware
//
#include <hip/hip_runtime.h>
#include <hip/hip_bf16.h>
#include <stdint.h>

typedef __attribute__((ext_vector_type(16))) __bf16 v16bf;
typedef __attribute__((ext_vector_type(8)))  float  v8f;
typedef unsigned short u16;

#define TSEQ  2048
#define CDIM  768
#define NHEAD 12
#define HDIM  64

__device__ __forceinline__ u16 f2bf(float f) {
  union { float f; unsigned u; } v; v.f = f;
  unsigned r = v.u + 0x7FFFu + ((v.u >> 16) & 1u);
  return (u16)(r >> 16);
}

// ---- CDNA5 async global->LDS copy (ASYNCcnt tracked, no VGPR staging) ------
__device__ __forceinline__ void async_b128(const void* lds_dst, const void* gsrc) {
  uint32_t lo = (uint32_t)(uintptr_t)lds_dst;      // low 32b of generic = LDS offset
  unsigned long long ga = (unsigned long long)(uintptr_t)gsrc;
  asm volatile("global_load_async_to_lds_b128 %0, %1, off"
               :: "v"(lo), "v"(ga) : "memory");
}
__device__ __forceinline__ void wait_async0() {
  asm volatile("s_wait_asynccnt 0x0" ::: "memory");
}

__global__ void cvt_f32_bf16(const float* __restrict__ in, u16* __restrict__ out, int n) {
  int i = blockIdx.x * blockDim.x + threadIdx.x;
  if (i < n) out[i] = f2bf(in[i]);
}

// ---------------------------------------------------------------------------
// GEMM: Y = X @ W^T + bias.  X:(M,K) bf16 row-major, W:(N,K) bf16 row-major.
// Block = 256 threads (8 waves). Tile M=128 (16 rows/wave), N=64, Kstep=32.
// Double-buffered LDS tiles filled by global_load_async_to_lds_b128 so the
// copy of step i+1 overlaps the WMMAs of step i.
// WMMA bf16 fragment layouts used here (A and B share one k order):
//   A lane: row M = lane&15; K run of 16 starting at 16*half
//   B lane: col N = lane&15; K run of 16 starting at 16*half
//   C lane: (M = r + 8*half, N = lane&15) for r = 0..7
// ---------------------------------------------------------------------------
template<bool F32OUT>
__global__ __launch_bounds__(256) void gemm_bf16wmma(
    const u16* __restrict__ X, const u16* __restrict__ W,
    const float* __restrict__ bias,
    u16* __restrict__ Yb, float* __restrict__ Yf,
    int M, int N, int K)
{
  __shared__ u16 sX[2][128 * 32];
  __shared__ u16 sW[2][64 * 32];
  const int tid = threadIdx.x, wave = tid >> 5, lane = tid & 31;
  const int l15 = lane & 15, half = lane >> 4;
  const int m0 = blockIdx.x * 128, n0 = blockIdx.y * 64;

  const int xrow = tid >> 1, xpart = tid & 1;      // 32B of X tile per thread
  const int wrow = tid >> 2, wpart = tid & 3;      // 16B of W tile per thread

  auto stage = [&](int buf, int kk) {
    const u16* gx = X + (size_t)(m0 + xrow) * K + kk + xpart * 16;
    async_b128(&sX[buf][xrow * 32 + xpart * 16],     gx);
    async_b128(&sX[buf][xrow * 32 + xpart * 16 + 8], gx + 8);
    const u16* gw = W + (size_t)(n0 + wrow) * K + kk + wpart * 8;
    async_b128(&sW[buf][wrow * 32 + wpart * 8], gw);
  };

  v8f zero = {0.f, 0.f, 0.f, 0.f, 0.f, 0.f, 0.f, 0.f};
  v8f acc[4];
#pragma unroll
  for (int t = 0; t < 4; ++t) acc[t] = zero;

  const int nk = K / 32;
  stage(0, 0);
  wait_async0();
  __syncthreads();

  for (int it = 0; it < nk; ++it) {
    const int buf = it & 1;
    if (it + 1 < nk) stage(buf ^ 1, (it + 1) * 32);   // overlap with compute

    union { v16bf v; uint4 q[2]; } a;
    const u16* ap = &sX[buf][(wave * 16 + l15) * 32 + 16 * half];
    a.q[0] = *(const uint4*)(ap);
    a.q[1] = *(const uint4*)(ap + 8);
#pragma unroll
    for (int t = 0; t < 4; ++t) {
      union { v16bf v; uint4 q[2]; } b;
      const u16* bp = &sW[buf][(t * 16 + l15) * 32 + 16 * half];
      b.q[0] = *(const uint4*)(bp);
      b.q[1] = *(const uint4*)(bp + 8);
      acc[t] = __builtin_amdgcn_wmma_f32_16x16x32_bf16(
          false, a.v, false, b.v, (short)0, acc[t], false, false);
    }
    wait_async0();       // next tile landed in LDS
    __syncthreads();     // all waves done reading buf; next tile visible
  }

#pragma unroll
  for (int t = 0; t < 4; ++t) {
    int col = n0 + t * 16 + l15;
    float bv = bias[col];
#pragma unroll
    for (int r = 0; r < 8; ++r) {
      int row = m0 + wave * 16 + r + 8 * half;
      float v = acc[t][r] + bv;
      if (F32OUT) Yf[(size_t)row * N + col] = v;
      else        Yb[(size_t)row * N + col] = f2bf(v);
    }
  }
}

// ---------------------------------------------------------------------------
// Fused flash-attention per (b, h, 128-query block). 8 waves x 16 q-rows.
// Key tiles of 64 staged in LDS: K (and mask-K) via async global->LDS,
// V transposed through VGPRs. S = Q K^T via WMMA; online softmax in fp32;
// P round-trips LDS to become an A-fragment; O += P V via WMMA.
// ---------------------------------------------------------------------------
template<bool HASMASK>
__global__ __launch_bounds__(256) void attn_wmma(
    const u16* __restrict__ Q, const u16* __restrict__ Kk,
    const u16* __restrict__ V, const u16* __restrict__ Mm,
    u16* __restrict__ O)
{
  __shared__ u16 sK [64 * 64];
  __shared__ u16 sVT[64 * 64];
  __shared__ u16 sM [64 * 64];
  __shared__ u16 sP [8 * 16 * 64];
  const int tid = threadIdx.x, wave = tid >> 5, lane = tid & 31;
  const int l15 = lane & 15, half = lane >> 4;
  const int q0 = blockIdx.x * 128, h = blockIdx.y, b = blockIdx.z;
  const size_t hoff = (size_t)h * HDIM;

  union { v16bf v; uint4 q[2]; } qa[2];
  {
    const u16* qp = Q + ((size_t)(b * TSEQ + q0 + wave * 16 + l15)) * CDIM + hoff;
#pragma unroll
    for (int s = 0; s < 2; ++s) {
      qa[s].q[0] = *(const uint4*)(qp + 32 * s + 16 * half);
      qa[s].q[1] = *(const uint4*)(qp + 32 * s + 16 * half + 8);
    }
  }
  v8f zero = {0.f, 0.f, 0.f, 0.f, 0.f, 0.f, 0.f, 0.f};
  v8f o[4];
#pragma unroll
  for (int t = 0; t < 4; ++t) o[t] = zero;
  float row_m[8], row_l[8];
#pragma unroll
  for (int r = 0; r < 8; ++r) { row_m[r] = -1e30f; row_l[r] = 0.f; }

  for (int kt = 0; kt < TSEQ / 64; ++kt) {
    __syncthreads();               // previous tile fully consumed
    {
      int row = tid >> 2, part = tid & 3;
      size_t gbase = ((size_t)(b * TSEQ + kt * 64 + row)) * CDIM + hoff + part * 16;
      // K (and mask) tiles: async straight into LDS
      async_b128(sK + row * 64 + part * 16,     Kk + gbase);
      async_b128(sK + row * 64 + part * 16 + 8, Kk + gbase + 8);
      if (HASMASK) {
        async_b128(sM + row * 64 + part * 16,     Mm + gbase);
        async_b128(sM + row * 64 + part * 16 + 8, Mm + gbase + 8);
      }
      // V tile: through VGPRs (needs transpose on the way into LDS)
      const uint4* vp = (const uint4*)(V + gbase);
      u16 tmp[16];
      *(uint4*)(tmp)     = vp[0];
      *(uint4*)(tmp + 8) = vp[1];
#pragma unroll
      for (int j = 0; j < 16; ++j) sVT[(part * 16 + j) * 64 + row] = tmp[j];
    }
    wait_async0();
    __syncthreads();

    // S = Q K^T (16 x 64 per wave)
    v8f s[4];
#pragma unroll
    for (int t = 0; t < 4; ++t) s[t] = zero;
#pragma unroll
    for (int t = 0; t < 4; ++t) {
#pragma unroll
      for (int s2 = 0; s2 < 2; ++s2) {
        union { v16bf v; uint4 q[2]; } bfr;
        const u16* bp = sK + (t * 16 + l15) * 64 + 32 * s2 + 16 * half;
        bfr.q[0] = *(const uint4*)bp;
        bfr.q[1] = *(const uint4*)(bp + 8);
        s[t] = __builtin_amdgcn_wmma_f32_16x16x32_bf16(
            false, qa[s2].v, false, bfr.v, (short)0, s[t], false, false);
      }
    }

    float sv[4][8];
    if (HASMASK) {
      v8f ms[4];
#pragma unroll
      for (int t = 0; t < 4; ++t) ms[t] = zero;
#pragma unroll
      for (int t = 0; t < 4; ++t) {
#pragma unroll
        for (int s2 = 0; s2 < 2; ++s2) {
          union { v16bf v; uint4 q[2]; } bfr;
          const u16* bp = sM + (t * 16 + l15) * 64 + 32 * s2 + 16 * half;
          bfr.q[0] = *(const uint4*)bp;
          bfr.q[1] = *(const uint4*)(bp + 8);
          ms[t] = __builtin_amdgcn_wmma_f32_16x16x32_bf16(
              false, qa[s2].v, false, bfr.v, (short)0, ms[t], false, false);
        }
      }
#pragma unroll
      for (int t = 0; t < 4; ++t)
#pragma unroll
        for (int r = 0; r < 8; ++r) {
          float x = s[t][r] * 0.125f;               // 1/sqrt(64)
          if (ms[t][r] != 0.f) x -= 10000.f;        // where(m!=0, 1, 0)*-1e4
          sv[t][r] = x;
        }
    } else {
#pragma unroll
      for (int t = 0; t < 4; ++t)
#pragma unroll
        for (int r = 0; r < 8; ++r) sv[t][r] = s[t][r] * 0.125f;
    }

    // online softmax: rows live on 16-lane halves -> xor-reduce 1,2,4,8
    float alpha[8];
#pragma unroll
    for (int r = 0; r < 8; ++r) {
      float mx = sv[0][r];
#pragma unroll
      for (int t = 1; t < 4; ++t) mx = fmaxf(mx, sv[t][r]);
#pragma unroll
      for (int d = 1; d < 16; d <<= 1) mx = fmaxf(mx, __shfl_xor(mx, d));
      float nm = fmaxf(row_m[r], mx);
      alpha[r] = __expf(row_m[r] - nm);
      row_m[r] = nm;
    }
    float lsum[8];
#pragma unroll
    for (int r = 0; r < 8; ++r) lsum[r] = 0.f;
    u16* pw = sP + wave * (16 * 64);
#pragma unroll
    for (int t = 0; t < 4; ++t)
#pragma unroll
      for (int r = 0; r < 8; ++r) {
        float p = __expf(sv[t][r] - row_m[r]);
        lsum[r] += p;
        pw[(r + 8 * half) * 64 + t * 16 + l15] = f2bf(p);
      }
#pragma unroll
    for (int r = 0; r < 8; ++r) {
      float ls = lsum[r];
#pragma unroll
      for (int d = 1; d < 16; d <<= 1) ls += __shfl_xor(ls, d);
      row_l[r] = row_l[r] * alpha[r] + ls;
    }
#pragma unroll
    for (int t = 0; t < 4; ++t)
#pragma unroll
      for (int r = 0; r < 8; ++r) o[t][r] *= alpha[r];
    __syncthreads();               // P visible / ordering for A-frag reload

    // O += P V
    union { v16bf v; uint4 q[2]; } pa[2];
#pragma unroll
    for (int s2 = 0; s2 < 2; ++s2) {
      const u16* pp = pw + l15 * 64 + 32 * s2 + 16 * half;
      pa[s2].q[0] = *(const uint4*)pp;
      pa[s2].q[1] = *(const uint4*)(pp + 8);
    }
#pragma unroll
    for (int t = 0; t < 4; ++t) {
#pragma unroll
      for (int s2 = 0; s2 < 2; ++s2) {
        union { v16bf v; uint4 q[2]; } vb;
        const u16* vp2 = sVT + (t * 16 + l15) * 64 + 32 * s2 + 16 * half;
        vb.q[0] = *(const uint4*)vp2;
        vb.q[1] = *(const uint4*)(vp2 + 8);
        o[t] = __builtin_amdgcn_wmma_f32_16x16x32_bf16(
            false, pa[s2].v, false, vb.v, (short)0, o[t], false, false);
      }
    }
  }

  // epilogue: O /= l, write merged (B*T, C) bf16
#pragma unroll
  for (int t = 0; t < 4; ++t)
#pragma unroll
    for (int r = 0; r < 8; ++r) {
      int row = q0 + wave * 16 + r + 8 * half;
      float val = o[t][r] / row_l[r];
      O[((size_t)(b * TSEQ + row)) * CDIM + hoff + t * 16 + l15] = f2bf(val);
    }
}

// ---------------------------------------------------------------------------
extern "C" void kernel_launch(void* const* d_in, const int* in_sizes, int n_in,
                              void* d_out, int out_size, void* d_ws, size_t ws_size,
                              hipStream_t stream) {
  const float* x_img  = (const float*)d_in[0];
  const float* y_bh   = (const float*)d_in[1];
  const float* amask  = (const float*)d_in[2];
  const float* Wv_i2b = (const float*)d_in[3];  const float* bv_i2b = (const float*)d_in[4];
  const float* Wk_i2b = (const float*)d_in[5];  const float* bk_i2b = (const float*)d_in[6];
  const float* Wq_img = (const float*)d_in[7];  const float* bq_img = (const float*)d_in[8];
  const float* Wv_b2i = (const float*)d_in[9];  const float* bv_b2i = (const float*)d_in[10];
  const float* Wk_b2i = (const float*)d_in[11]; const float* bk_b2i = (const float*)d_in[12];
  const float* Wq_bh  = (const float*)d_in[13]; const float* bq_bh  = (const float*)d_in[14];
  const float* Wp_img = (const float*)d_in[15]; const float* bp_img = (const float*)d_in[16];
  const float* Wp_bh  = (const float*)d_in[17]; const float* bp_bh  = (const float*)d_in[18];

  const size_t ACT = (size_t)2 * TSEQ * CDIM;   // 3,145,728
  const size_t WEL = (size_t)CDIM * CDIM;       //   589,824
  u16* ws  = (u16*)d_ws;
  u16* xb  = ws;
  u16* yb  = xb  + ACT;
  u16* mb  = yb  + ACT;
  u16* wq_i = mb   + ACT;
  u16* wk_i = wq_i + WEL;
  u16* wv_i = wk_i + WEL;
  u16* wq_b = wv_i + WEL;
  u16* wk_b = wq_b + WEL;
  u16* wv_b = wk_b + WEL;
  u16* wp_i = wv_b + WEL;
  u16* wp_b = wp_i + WEL;
  u16* qib = wp_b + WEL;
  u16* kib = qib + ACT;
  u16* vib = kib + ACT;
  u16* qbb = vib + ACT;
  u16* kbb = qbb + ACT;
  u16* vbb = kbb + ACT;
  u16* aoi = vbb + ACT;
  u16* aob = aoi + ACT;

  const int cb = 256;
  auto cv = [&](const float* s, u16* d, size_t n) {
    cvt_f32_bf16<<<dim3((unsigned)((n + cb - 1) / cb)), dim3(cb), 0, stream>>>(s, d, (int)n);
  };
  cv(x_img, xb, ACT);  cv(y_bh, yb, ACT);  cv(amask, mb, ACT);
  cv(Wq_img, wq_i, WEL); cv(Wk_i2b, wk_i, WEL); cv(Wv_i2b, wv_i, WEL);
  cv(Wq_bh,  wq_b, WEL); cv(Wk_b2i, wk_b, WEL); cv(Wv_b2i, wv_b, WEL);
  cv(Wp_img, wp_i, WEL); cv(Wp_bh,  wp_b, WEL);

  const int M = 2 * TSEQ;                       // 4096 rows
  dim3 gg(M / 128, CDIM / 64);                  // 32 x 12
  gemm_bf16wmma<false><<<gg, 256, 0, stream>>>(xb, wq_i, bq_img, qib, nullptr, M, CDIM, CDIM);
  gemm_bf16wmma<false><<<gg, 256, 0, stream>>>(xb, wk_i, bk_i2b, kib, nullptr, M, CDIM, CDIM);
  gemm_bf16wmma<false><<<gg, 256, 0, stream>>>(xb, wv_i, bv_i2b, vib, nullptr, M, CDIM, CDIM);
  gemm_bf16wmma<false><<<gg, 256, 0, stream>>>(yb, wq_b, bq_bh,  qbb, nullptr, M, CDIM, CDIM);
  gemm_bf16wmma<false><<<gg, 256, 0, stream>>>(yb, wk_b, bk_b2i, kbb, nullptr, M, CDIM, CDIM);
  gemm_bf16wmma<false><<<gg, 256, 0, stream>>>(yb, wv_b, bv_b2i, vbb, nullptr, M, CDIM, CDIM);

  dim3 ga(TSEQ / 128, NHEAD, 2);                // 16 x 12 x 2 = 384 WGs
  attn_wmma<true ><<<ga, 256, 0, stream>>>(qib, kbb, vbb, mb, aoi);
  attn_wmma<false><<<ga, 256, 0, stream>>>(qbb, kib, vib, nullptr, aob);

  float* out_img = (float*)d_out;
  float* out_bh  = out_img + ACT;
  gemm_bf16wmma<true><<<gg, 256, 0, stream>>>(aoi, wp_i, bp_img, nullptr, out_img, M, CDIM, CDIM);
  gemm_bf16wmma<true><<<gg, 256, 0, stream>>>(aob, wp_b, bp_bh,  nullptr, out_bh,  M, CDIM, CDIM);
}
